// KNNDensityEstimator_2594160247093
// MI455X (gfx1250) — matrix operand order repaired and checked by reading, not compiled, not run on hardware
//
#include <hip/hip_runtime.h>
#include <hip/hip_bf16.h>
#include <float.h>

typedef __attribute__((ext_vector_type(16))) _Float16 v16h;
typedef __attribute__((ext_vector_type(8)))  _Float16 v8h;
typedef __attribute__((ext_vector_type(4)))  _Float16 v4h;
typedef __attribute__((ext_vector_type(8)))  float    v8f;
typedef __attribute__((ext_vector_type(4)))  float    v4f;

#define KNN_K   10
#define DIMS    128
#define SLICES  16
#define NWAVES  8

// Split f32 -> f16 hi + f16 lo residual (Ootomo-style precision recovery).
// Macro because clang forbids taking the address of an ext_vector element.
#define SPLIT_TO(vecH, vecL, idx, fval)                            \
  do {                                                             \
    float _f = (fval);                                             \
    _Float16 _h = (_Float16)_f;                                    \
    (vecH)[(idx)] = _h;                                            \
    (vecL)[(idx)] = (_Float16)(_f - (float)_h);                    \
  } while (0)

__device__ __forceinline__ void topk_insert(float (&arr)[KNN_K], float v) {
  // Invariant: arr[0] holds the max of the kept K values.
  if (v < arr[0]) {
    arr[0] = v;
#pragma unroll
    for (int j = 1; j < KNN_K; ++j) {
      if (arr[j] > arr[0]) { float t = arr[0]; arr[0] = arr[j]; arr[j] = t; }
    }
  }
}

// ---------------------------------------------------------------------------
// Kernel 0: one-time f32 -> (f16 hi, f16 lo) split of ref_feats, layout
// preserved ([N][128] row-major) so WMMA B-fragment loads stay contiguous.
// ---------------------------------------------------------------------------
__global__ __launch_bounds__(256)
void split_convert_kernel(const float* __restrict__ src,
                          _Float16* __restrict__ hi,
                          _Float16* __restrict__ lo,
                          long long n4) {  // n/4
  const long long i = (long long)blockIdx.x * blockDim.x + threadIdx.x;
  if (i >= n4) return;
  const v4f f = *(const v4f*)(src + i * 4);
  v4h h, l;
#pragma unroll
  for (int j = 0; j < 4; ++j) SPLIT_TO(h, l, j, f[j]);
  *(v4h*)(hi + i * 4) = h;
  *(v4h*)(lo + i * 4) = l;
}

// ---------------------------------------------------------------------------
// Kernel 1: squared row norms, one wave32 per row of 128 floats.
// ---------------------------------------------------------------------------
__global__ __launch_bounds__(256)
void rownorm_kernel(const float* __restrict__ src, float* __restrict__ out,
                    int nrows) {
  const int gid  = blockIdx.x * blockDim.x + threadIdx.x;
  const int row  = gid >> 5;
  const int lane = gid & 31;
  if (row >= nrows) return;
  const v4f v = *(const v4f*)(src + (size_t)row * DIMS + lane * 4);
  float s = v[0] * v[0] + v[1] * v[1] + v[2] * v[2] + v[3] * v[3];
#pragma unroll
  for (int off = 16; off >= 1; off >>= 1) s += __shfl_xor(s, off, 32);
  if (lane == 0) out[row] = s;
}

// ---------------------------------------------------------------------------
// Kernel 2: per (16-row tile, N-slice): stream 16-wide column tiles through
// v_wmma_f32_16x16x32_f16 (hi/lo split for ~f32 accuracy), per-lane register
// top-10 of d2, block-wide LDS merge, emit 10 candidates/row/slice.
// Partial tiles handled by CLAMPING the column (garbage-but-finite dot,
// forced d2 ~ FLT_MAX via cndmask) so the hot loop is fully branch-free.
// Three independent WMMA accumulator chains give ILP across the XDL pipe.
// ---------------------------------------------------------------------------
template <bool PRESPLIT>
__global__ __launch_bounds__(256)
void knn_tile_kernel(const float* __restrict__ feat,
                     const float* __restrict__ ref,
                     const _Float16* __restrict__ refh,
                     const _Float16* __restrict__ refl,
                     const float* __restrict__ x2,
                     const float* __restrict__ y2,
                     float* __restrict__ cand,
                     int N, int sliceLen) {
  // Two-pass merge keeps static LDS at 40KB + 10KB = 50KB.
  __shared__ float ldsCand[4 * 16 * 16 * KNN_K];
  __shared__ float lds2[16 * 16 * KNN_K];

  const int tid  = threadIdx.x;
  const int wave = tid >> 5;
  const int lane = tid & 31;
  const int half = lane >> 4;
  const int l    = lane & 15;

  const int rowBase  = blockIdx.x * 16;
  const int slice    = blockIdx.y;
  const int sliceBeg = slice * sliceLen;
  const int sliceEnd = min(sliceBeg + sliceLen, N);
  const int sliceCnt = sliceEnd - sliceBeg;

  // --- Preload A fragments (16 rows x 128 K), hi/lo split, 4 K-chunks.
  // A 16x32 layout (matches B): lane row m = lane&15;
  //   i in [0,16): k = kb + 16*half + i
  v16h ah[4], al[4];
  {
    const float* fp = feat + (size_t)(rowBase + l) * DIMS;
#pragma unroll
    for (int c = 0; c < 4; ++c) {
      const int kb = c * 32;
      const v4f f0 = *(const v4f*)(fp + kb + 16 * half);
      const v4f f1 = *(const v4f*)(fp + kb + 16 * half + 4);
      const v4f f2 = *(const v4f*)(fp + kb + 16 * half + 8);
      const v4f f3 = *(const v4f*)(fp + kb + 16 * half + 12);
#pragma unroll
      for (int i = 0; i < 4; ++i) {
        SPLIT_TO(ah[c], al[c], i,      f0[i]);
        SPLIT_TO(ah[c], al[c], 4 + i,  f1[i]);
        SPLIT_TO(ah[c], al[c], 8 + i,  f2[i]);
        SPLIT_TO(ah[c], al[c], 12 + i, f3[i]);
      }
    }
  }

  // C layout: VGPR r holds row m = r + 8*half.
  float x2row[8];
#pragma unroll
  for (int r = 0; r < 8; ++r) x2row[r] = x2[rowBase + r + 8 * half];

  float top[8][KNN_K];
#pragma unroll
  for (int r = 0; r < 8; ++r)
#pragma unroll
    for (int j = 0; j < KNN_K; ++j) top[r][j] = FLT_MAX;

  const v8f z8 = {0.f, 0.f, 0.f, 0.f, 0.f, 0.f, 0.f, 0.f};
  const int numTiles = (sliceCnt + 15) / 16;

  for (int t = wave; t < numTiles; t += NWAVES) {
    const int colRaw = sliceBeg + t * 16 + l;  // B layout: lane col n = lane&15
    const bool valid = colRaw < sliceEnd;
    const int col    = valid ? colRaw : (sliceEnd - 1);   // clamp, branch-free
    const int colPf  = min(colRaw + NWAVES * 16, sliceEnd - 1);

    v8f a0, a1, a2;  // ah*bh, ah*bl, al*bh chains

    if constexpr (PRESPLIT) {
      const _Float16* bph = refh + (size_t)col * DIMS + 16 * half;
      const _Float16* bpl = refl + (size_t)col * DIMS + 16 * half;
      // locality=3 -> WGP scope: pull into all cache levels (stream is
      // L2-resident; this hides L2 -> WGP latency). Address clamped in-bounds.
      __builtin_prefetch(refh + (size_t)colPf * DIMS + 16 * half, 0, 3);
      __builtin_prefetch(refl + (size_t)colPf * DIMS + 16 * half, 0, 3);
#pragma unroll
      for (int cc = 0; cc < 4; ++cc) {
        // B 32x16 order used here (same as A): k = kb + 16*half + i -> 16 contiguous f16.
        const v8h h0 = *(const v8h*)(bph + cc * 32);
        const v8h h1 = *(const v8h*)(bph + cc * 32 + 8);
        const v8h l0 = *(const v8h*)(bpl + cc * 32);
        const v8h l1 = *(const v8h*)(bpl + cc * 32 + 8);
        const v16h bh = __builtin_shufflevector(h0, h1, 0, 1, 2, 3, 4, 5, 6, 7,
                                                8, 9, 10, 11, 12, 13, 14, 15);
        const v16h bl = __builtin_shufflevector(l0, l1, 0, 1, 2, 3, 4, 5, 6, 7,
                                                8, 9, 10, 11, 12, 13, 14, 15);
        a0 = __builtin_amdgcn_wmma_f32_16x16x32_f16(false, ah[cc], false, bh,
                                                    (short)0, cc ? a0 : z8,
                                                    false, false);
        a1 = __builtin_amdgcn_wmma_f32_16x16x32_f16(false, ah[cc], false, bl,
                                                    (short)0, cc ? a1 : z8,
                                                    false, false);
        a2 = __builtin_amdgcn_wmma_f32_16x16x32_f16(false, al[cc], false, bh,
                                                    (short)0, cc ? a2 : z8,
                                                    false, false);
      }
    } else {
      const float* bp = ref + (size_t)col * DIMS + 16 * half;
      __builtin_prefetch(ref + (size_t)colPf * DIMS + 16 * half, 0, 3);
#pragma unroll
      for (int cc = 0; cc < 4; ++cc) {
        const v4f* q = (const v4f*)(bp + cc * 32);
        const v4f b0 = q[0], b1 = q[1], b2 = q[2], b3 = q[3];
        v16h bh, bl;
#pragma unroll
        for (int i = 0; i < 4; ++i) {
          SPLIT_TO(bh, bl, i,      b0[i]);
          SPLIT_TO(bh, bl, 4 + i,  b1[i]);
          SPLIT_TO(bh, bl, 8 + i,  b2[i]);
          SPLIT_TO(bh, bl, 12 + i, b3[i]);
        }
        a0 = __builtin_amdgcn_wmma_f32_16x16x32_f16(false, ah[cc], false, bh,
                                                    (short)0, cc ? a0 : z8,
                                                    false, false);
        a1 = __builtin_amdgcn_wmma_f32_16x16x32_f16(false, ah[cc], false, bl,
                                                    (short)0, cc ? a1 : z8,
                                                    false, false);
        a2 = __builtin_amdgcn_wmma_f32_16x16x32_f16(false, al[cc], false, bh,
                                                    (short)0, cc ? a2 : z8,
                                                    false, false);
      }
    }

    // Unconditional (clamped) load + select: no exec-mask branch in the loop.
    const float y2v = y2[col];
    const float y2n = valid ? y2v : FLT_MAX;
#pragma unroll
    for (int r = 0; r < 8; ++r) {
      const float dot = a0[r] + (a1[r] + a2[r]);
      const float d2  = x2row[r] + y2n - 2.0f * dot;
      topk_insert(top[r], d2);
    }
  }

  // --- Merge: per-wave register lists -> LDS -> 10 candidates per row.
  float loc[KNN_K];
#pragma unroll
  for (int j = 0; j < KNN_K; ++j) loc[j] = FLT_MAX;

  for (int pass = 0; pass < 2; ++pass) {
    if ((wave >> 2) == pass) {
      const int w = wave & 3;
#pragma unroll
      for (int r = 0; r < 8; ++r) {
        const int m = r + 8 * half;
        float* dst = &ldsCand[(((w * 16) + m) * 16 + l) * KNN_K];
#pragma unroll
        for (int j = 0; j < KNN_K; ++j) dst[j] = top[r][j];
      }
    }
    __syncthreads();
    {
      const int m = tid >> 4;   // row 0..15
      const int p = tid & 15;   // partition 0..15
      for (int g = p; g < 64; g += 16) {
        const int w  = g >> 4;
        const int ll = g & 15;
        const float* src = &ldsCand[(((w * 16) + m) * 16 + ll) * KNN_K];
#pragma unroll
        for (int j = 0; j < KNN_K; ++j) topk_insert(loc, src[j]);
      }
    }
    __syncthreads();
  }

  {
    const int m = tid >> 4;
    const int p = tid & 15;
    float* dst = &lds2[(m * 16 + p) * KNN_K];
#pragma unroll
    for (int j = 0; j < KNN_K; ++j) dst[j] = loc[j];
  }
  __syncthreads();

  if (tid < 16) {
    const int m = tid;
    float fin[KNN_K];
#pragma unroll
    for (int j = 0; j < KNN_K; ++j) fin[j] = FLT_MAX;
    const float* src = &lds2[m * 16 * KNN_K];
    for (int i = 0; i < 16 * KNN_K; ++i) topk_insert(fin, src[i]);
    float* dst = cand + ((size_t)(rowBase + m) * SLICES + slice) * KNN_K;
#pragma unroll
    for (int j = 0; j < KNN_K; ++j) dst[j] = fin[j];
  }
}

// ---------------------------------------------------------------------------
// Kernel 3: merge SLICES*K candidates per row -> density.
// ---------------------------------------------------------------------------
__global__ __launch_bounds__(256)
void knn_final_kernel(const float* __restrict__ cand, float* __restrict__ out,
                      int rows) {
  const int row = blockIdx.x * blockDim.x + threadIdx.x;
  if (row >= rows) return;
  float fin[KNN_K];
#pragma unroll
  for (int j = 0; j < KNN_K; ++j) fin[j] = FLT_MAX;
  const float* src = cand + (size_t)row * SLICES * KNN_K;
  for (int i = 0; i < SLICES * KNN_K; ++i) topk_insert(fin, src[i]);
  float s = 0.f;
#pragma unroll
  for (int j = 0; j < KNN_K; ++j) s += sqrtf(fmaxf(fin[j], 0.0f));
  out[row] = -s / (float)KNN_K;
}

// ---------------------------------------------------------------------------
extern "C" void kernel_launch(void* const* d_in, const int* in_sizes, int n_in,
                              void* d_out, int out_size, void* d_ws,
                              size_t ws_size, hipStream_t stream) {
  const float* feat = (const float*)d_in[0];
  const float* ref  = (const float*)d_in[1];
  float* out        = (float*)d_out;

  const int Brows = in_sizes[0] / DIMS;  // 512
  const int N     = in_sizes[1] / DIMS;  // 100000

  // Workspace layout (256B aligned): y2 | x2 | cand | refh | refl
  char* ws   = (char*)d_ws;
  size_t off = 0;
  float* y2 = (float*)(ws + off);
  off += ((size_t)N * sizeof(float) + 255) & ~(size_t)255;
  float* x2 = (float*)(ws + off);
  off += ((size_t)Brows * sizeof(float) + 255) & ~(size_t)255;
  float* cand = (float*)(ws + off);
  off += ((size_t)Brows * SLICES * KNN_K * sizeof(float) + 255) & ~(size_t)255;
  _Float16* refh = (_Float16*)(ws + off);
  off += ((size_t)N * DIMS * sizeof(_Float16) + 255) & ~(size_t)255;
  _Float16* refl = (_Float16*)(ws + off);
  off += ((size_t)N * DIMS * sizeof(_Float16) + 255) & ~(size_t)255;
  const size_t bigNeed = off;

  const bool presplit = (ws_size >= bigNeed);

  // 1) row norms (one wave per row)
  {
    const long long threads = (long long)N * 32;
    rownorm_kernel<<<(int)((threads + 255) / 256), 256, 0, stream>>>(ref, y2, N);
  }
  {
    const long long threads = (long long)Brows * 32;
    rownorm_kernel<<<(int)((threads + 255) / 256), 256, 0, stream>>>(feat, x2,
                                                                     Brows);
  }

  const int sliceLen = (N + SLICES - 1) / SLICES;
  dim3 grid(Brows / 16, SLICES);

  if (presplit) {
    // 1b) one-time hi/lo split of ref_feats into f16 workspace buffers
    const long long n4 = (long long)N * DIMS / 4;
    split_convert_kernel<<<(int)((n4 + 255) / 256), 256, 0, stream>>>(
        ref, refh, refl, n4);
    // 2) tiled WMMA distance + per-slice top-K (conversion-free hot loop)
    knn_tile_kernel<true><<<grid, 256, 0, stream>>>(feat, ref, refh, refl, x2,
                                                    y2, cand, N, sliceLen);
  } else {
    knn_tile_kernel<false><<<grid, 256, 0, stream>>>(feat, ref, refh, refl, x2,
                                                     y2, cand, N, sliceLen);
  }

  // 3) final merge + density
  knn_final_kernel<<<(Brows + 255) / 256, 256, 0, stream>>>(cand, out, Brows);
}
